// MultiHeadAttention_35742717837571
// MI455X (gfx1250) — matrix operand order repaired and checked by reading, not compiled, not run on hardware
//
#include <hip/hip_runtime.h>

#define EMBED 1024
#define NHEAD 16
#define HDIM  64
#define BATCH 2
#define SEQ   2048
#define MROWS (BATCH * SEQ) /* 4096 */

// LDS panel: 64 rows x 64 bf16, TDM-padded by 4 DWORDs every 32 DWORDs
// -> row stride 72 elements (144B), panel 64*72 = 4608 elems = 9216B
#define PANEL_ROW   72
#define PANEL_ELEMS (64 * PANEL_ROW)
#define PANEL_BYTES (PANEL_ELEMS * 2)
#define NPANEL      (EMBED / 64)

typedef __attribute__((ext_vector_type(16))) __bf16         v16bf;
typedef __attribute__((ext_vector_type(8)))  float          v8f;
typedef __attribute__((ext_vector_type(8)))  unsigned short v8u;
typedef __attribute__((ext_vector_type(16))) unsigned short v16u;
typedef __attribute__((ext_vector_type(4)))  unsigned int   u32x4;
typedef __attribute__((ext_vector_type(8)))  int            i32x8;
typedef __attribute__((ext_vector_type(4)))  int            i32x4;
typedef unsigned short bfraw;

static __device__ __forceinline__ unsigned short f2bfu(float f) {
    unsigned int u = __builtin_bit_cast(unsigned int, f);
    u += 0x7FFFu + ((u >> 16) & 1u);          // round-to-nearest-even
    return (unsigned short)(u >> 16);
}

static __device__ __forceinline__ v8f v8f_zero() {
    v8f z;
#pragma unroll
    for (int i = 0; i < 8; ++i) z[i] = 0.0f;
    return z;
}

static __device__ __forceinline__ v16bf ldfrag2(const bfraw* p0, const bfraw* p1) {
    v8u lo = *(const v8u*)p0;
    v8u hi = *(const v8u*)p1;
    v16u t = __builtin_shufflevector(lo, hi, 0,1,2,3,4,5,6,7,8,9,10,11,12,13,14,15);
    return __builtin_bit_cast(v16bf, t);
}

// A-matrix fragment (16xK row-major tile; rowp = &M[row][kbase]).
// Lanes 0-15: K = {0..7, 16..23}; lanes 16-31: K = {8..15, 24..31}.
static __device__ __forceinline__ v16bf ldA(const bfraw* rowp, int hl) {
    return ldfrag2(rowp + hl * 8, rowp + hl * 8 + 16);
}
// B-matrix fragment (KxN; column n contiguous in memory; rowp = &Mem[n][kbase]).
// Lanes 0-15: K = {0..7, 16..23}; lanes 16-31: K = {8..15, 24..31}.
static __device__ __forceinline__ v16bf ldB(const bfraw* rowp, int hl) {
    return ldfrag2(rowp + hl * 8, rowp + hl * 8 + 16);
}

static __device__ __forceinline__ v8f wmma_bf16(v16bf a, v16bf b, v8f c) {
    return __builtin_amdgcn_wmma_f32_16x16x32_bf16(false, a, false, b, (short)0, c,
                                                   false, false);
}

// ---- Tensor Data Mover: async-load one 64(k) x 64(rows) bf16 panel into LDS.
// D# per CDNA5 ISA ch.8: group0 {count,lds_addr,global_addr,type=2},
// group1 {data_size=2B, pad_enable, pad_interval=32dw, pad_amount=4dw,
//         tensor dims, tile 64x64, dim0 stride}.
static __device__ __forceinline__ void tdm_load_panel(const bfraw* gsrc,
                                                      unsigned lds_byte_off,
                                                      unsigned row_stride_elems) {
    unsigned long long ga = (unsigned long long)(const void*)gsrc;
    u32x4 g0;
    g0[0] = 1u;                                              // count=1 (valid)
    g0[1] = lds_byte_off;                                    // lds_addr
    g0[2] = (unsigned)ga;                                    // global_addr[31:0]
    g0[3] = (unsigned)((ga >> 32) & 0x01FFFFFFull) | 0x80000000u; // addr[56:32]|type=2

    const unsigned td0 = 1u << 20, td1 = 1u << 20;           // generous tensor dims
    i32x8 g1;
    g1[0] = (int)((1u << 16)        // data_size = 2 bytes
                | (1u << 20)        // pad_enable
                | (4u << 22)        // pad_interval: 32 DWORDs
                | (3u << 25));      // pad_amount: 4 DWORDs
    g1[1] = (int)((td0 & 0xFFFFu) << 16);                            // dim0[15:0]
    g1[2] = (int)(((td0 >> 16) & 0xFFFFu) | ((td1 & 0xFFFFu) << 16)); // dim0[31:16]|dim1[15:0]
    g1[3] = (int)(((td1 >> 16) & 0xFFFFu) | (64u << 16));             // dim1[31:16]|tile_dim0=64
    g1[4] = (int)64u;                                                 // tile_dim1=64, tile_dim2=0
    g1[5] = (int)row_stride_elems;                                    // dim0_stride[31:0]
    g1[6] = 0;                                                        // stride hi / dim1_stride lo
    g1[7] = 0;
    i32x4 z4;
    z4[0] = 0; z4[1] = 0; z4[2] = 0; z4[3] = 0;
    i32x8 z8;
#pragma unroll
    for (int i = 0; i < 8; ++i) z8[i] = 0;
    // 6-arg form (clang-23 / therock HIP headers)
    __builtin_amdgcn_tensor_load_to_lds(g0, g1, z4, z4, z8, 0);
}

// ---------------------------------------------------------------- fp32 -> bf16
__global__ void cvt_bf16_kernel(const float* __restrict__ src,
                                bfraw* __restrict__ dst, int n) {
    int i = blockIdx.x * blockDim.x + threadIdx.x;
    int stride = gridDim.x * blockDim.x;
    for (; i < n; i += stride) dst[i] = f2bfu(src[i]);
}

// ------------------------------------------------- QKV projection: y = x*W^T+b
// out layout: vt_mode==0 -> [b,h,s,hd]; vt_mode==1 -> V transposed [b,h,hd,s]
__global__ __launch_bounds__(256) void qkv_proj_kernel(
    const bfraw* __restrict__ X, const bfraw* __restrict__ W,
    const float* __restrict__ bias, bfraw* __restrict__ out, int vt_mode) {
    extern __shared__ bfraw smem[];            // 2 * PANEL_BYTES
    int lane = threadIdx.x & 31;
    int wave = threadIdx.x >> 5;
    int hl = lane >> 4;
    int ln = lane & 15;
    int mbase = blockIdx.x * 128 + wave * 16;
    int nbase = blockIdx.y * 64;

    const bfraw* wpanel = W + (size_t)nbase * EMBED;  // 64 weight rows
    if (threadIdx.x == 0) tdm_load_panel(wpanel, 0u, EMBED);

    v8f acc[4];
#pragma unroll
    for (int j = 0; j < 4; ++j) acc[j] = v8f_zero();

    const bfraw* arow = X + (size_t)(mbase + ln) * EMBED;

    for (int p = 0; p < NPANEL; ++p) {
        int kp = p * 64;
        if (threadIdx.x == 0) {
            if (p + 1 < NPANEL) {
                tdm_load_panel(wpanel + (kp + 64),
                               (unsigned)(((p + 1) & 1) * PANEL_BYTES), EMBED);
                __builtin_amdgcn_s_wait_tensorcnt(1);   // panel p landed
            } else {
                __builtin_amdgcn_s_wait_tensorcnt(0);
            }
        }
        __syncthreads();
        const bfraw* bpan = smem + (p & 1) * PANEL_ELEMS;
        __builtin_prefetch(arow + kp + 128, 0, 3);
#pragma unroll
        for (int kk = 0; kk < 64; kk += 32) {
            v16bf a = ldA(arow + kp + kk, hl);
#pragma unroll
            for (int j = 0; j < 4; ++j) {
                v16bf b = ldB(bpan + (16 * j + ln) * PANEL_ROW + kk, hl);
                acc[j] = wmma_bf16(a, b, acc[j]);
            }
        }
        __syncthreads();
    }

#pragma unroll
    for (int j = 0; j < 4; ++j) {
        int n = nbase + 16 * j + ln;
        float bn = bias[n];
        int hh = n >> 6, e = n & (HDIM - 1);
#pragma unroll
        for (int i = 0; i < 8; ++i) {
            int m = mbase + i + hl * 8;
            int bb = m >> 11, s = m & (SEQ - 1);
            size_t addr = vt_mode
                ? ((size_t)((bb * NHEAD + hh) * HDIM + e) * SEQ + s)
                : ((size_t)((bb * NHEAD + hh) * SEQ + s) * HDIM + e);
            out[addr] = f2bfu(acc[j][i] + bn);
        }
    }
}

// ----------------------------------------------- causal flash attention (bf16)
// Q,K: [b,h,S,64]; Vt: [b,h,64,S]; AO: [b,s,h*64+d] bf16
__global__ __launch_bounds__(256) void flash_attn_kernel(
    const bfraw* __restrict__ Qb, const bfraw* __restrict__ Kb,
    const bfraw* __restrict__ Vt, bfraw* __restrict__ AO) {
    int lane = threadIdx.x & 31;
    int wave = threadIdx.x >> 5;
    int hl = lane >> 4;
    int ln = lane & 15;
    int b = blockIdx.z, h = blockIdx.y;
    int qbase = blockIdx.x * 128 + wave * 16;

    const bfraw* Qh = Qb + (size_t)(b * NHEAD + h) * SEQ * HDIM;
    const bfraw* Kh = Kb + (size_t)(b * NHEAD + h) * SEQ * HDIM;
    const bfraw* Vh = Vt + (size_t)(b * NHEAD + h) * HDIM * SEQ;
    const float scale = 0.125f; // 1/sqrt(64)

    v8f o[4];
#pragma unroll
    for (int j = 0; j < 4; ++j) o[j] = v8f_zero();
    float rmax = -1e30f, rsum = 0.0f;

    int qg = qbase + ln;                     // this lane's query column
    const bfraw* qrow = Qh + (size_t)qg * HDIM;
    v16bf bq0 = ldB(qrow, hl);               // d 0..31
    v16bf bq1 = ldB(qrow + 32, hl);          // d 32..63

    int ntiles = (qbase + 47) >> 5;          // causal key-tile count (32 keys/tile)
    for (int t = 0; t < ntiles; ++t) {
        int kb = t * 32;
        const bfraw* kr0 = Kh + (size_t)(kb + ln) * HDIM;  // keys kb..kb+15
        const bfraw* kr1 = kr0 + 16 * HDIM;                // keys kb+16..kb+31

        // S^T = K_tile * Q^T  (lane = q column; rows = keys)
        v8f st0 = v8f_zero(), st1 = v8f_zero();
        st0 = wmma_bf16(ldA(kr0, hl), bq0, st0);
        st0 = wmma_bf16(ldA(kr0 + 32, hl), bq1, st0);
        st1 = wmma_bf16(ldA(kr1, hl), bq0, st1);
        st1 = wmma_bf16(ldA(kr1 + 32, hl), bq1, st1);

        // scale + causal mask + tile max
        float tmax = -1e30f;
#pragma unroll
        for (int i = 0; i < 8; ++i) {
            int k0 = kb + i + hl * 8;
            float s0 = st0[i] * scale;
            if (k0 > qg) s0 = -1e30f;
            float s1 = st1[i] * scale;
            if (k0 + 16 > qg) s1 = -1e30f;
            st0[i] = s0;
            st1[i] = s1;
            tmax = fmaxf(tmax, fmaxf(s0, s1));
        }
        tmax = fmaxf(tmax, __shfl_xor(tmax, 16, 32));
        float mnew = fmaxf(rmax, tmax);
        float alpha = __expf(rmax - mnew);

        // P^T = exp(S^T - m); pack straight into the A-fragment layout for P@V
        float tsum = 0.0f;
        v16u pu;
#pragma unroll
        for (int i = 0; i < 8; ++i) {
            float p0 = __expf(st0[i] - mnew);
            float p1 = __expf(st1[i] - mnew);
            tsum += p0 + p1;
            pu[i] = f2bfu(p0);
            pu[i + 8] = f2bfu(p1);
        }
        tsum += __shfl_xor(tsum, 16, 32);
        rsum = rsum * alpha + tsum;
        rmax = mnew;
        v16bf pa = __builtin_bit_cast(v16bf, pu);

        // rescale O by per-row alpha (broadcast q-row stats into O layout)
#pragma unroll
        for (int i = 0; i < 8; ++i) {
            float ai = __shfl(alpha, i + hl * 8, 32);
            o[0][i] *= ai; o[1][i] *= ai; o[2][i] *= ai; o[3][i] *= ai;
        }

        // O += P * V  (B-fragments from pre-transposed V: contiguous key runs)
#pragma unroll
        for (int j = 0; j < 4; ++j) {
            const bfraw* vr = Vh + (size_t)(16 * j + ln) * SEQ + kb;
            o[j] = wmma_bf16(pa, ldB(vr, hl), o[j]);
        }
    }

    // normalize by row sums
#pragma unroll
    for (int i = 0; i < 8; ++i) {
        float rs = __shfl(rsum, i + hl * 8, 32);
        float inv = 1.0f / rs;
        o[0][i] *= inv; o[1][i] *= inv; o[2][i] *= inv; o[3][i] *= inv;
    }

    // store AO[b, s, h*64+d] bf16
#pragma unroll
    for (int j = 0; j < 4; ++j) {
        int d = 16 * j + ln;
#pragma unroll
        for (int i = 0; i < 8; ++i) {
            int s = qbase + i + hl * 8;
            AO[(size_t)(b * SEQ + s) * EMBED + h * HDIM + d] = f2bfu(o[j][i]);
        }
    }
}

// -------------------------------------------- output projection -> fp32 d_out
__global__ __launch_bounds__(256) void out_proj_kernel(
    const bfraw* __restrict__ X, const bfraw* __restrict__ W,
    const float* __restrict__ bias, float* __restrict__ out) {
    extern __shared__ bfraw smem[];
    int lane = threadIdx.x & 31;
    int wave = threadIdx.x >> 5;
    int hl = lane >> 4;
    int ln = lane & 15;
    int mbase = blockIdx.x * 128 + wave * 16;
    int nbase = blockIdx.y * 64;

    const bfraw* wpanel = W + (size_t)nbase * EMBED;
    if (threadIdx.x == 0) tdm_load_panel(wpanel, 0u, EMBED);

    v8f acc[4];
#pragma unroll
    for (int j = 0; j < 4; ++j) acc[j] = v8f_zero();

    const bfraw* arow = X + (size_t)(mbase + ln) * EMBED;

    for (int p = 0; p < NPANEL; ++p) {
        int kp = p * 64;
        if (threadIdx.x == 0) {
            if (p + 1 < NPANEL) {
                tdm_load_panel(wpanel + (kp + 64),
                               (unsigned)(((p + 1) & 1) * PANEL_BYTES), EMBED);
                __builtin_amdgcn_s_wait_tensorcnt(1);
            } else {
                __builtin_amdgcn_s_wait_tensorcnt(0);
            }
        }
        __syncthreads();
        const bfraw* bpan = smem + (p & 1) * PANEL_ELEMS;
        __builtin_prefetch(arow + kp + 128, 0, 3);
#pragma unroll
        for (int kk = 0; kk < 64; kk += 32) {
            v16bf a = ldA(arow + kp + kk, hl);
#pragma unroll
            for (int j = 0; j < 4; ++j) {
                v16bf b = ldB(bpan + (16 * j + ln) * PANEL_ROW + kk, hl);
                acc[j] = wmma_bf16(a, b, acc[j]);
            }
        }
        __syncthreads();
    }

#pragma unroll
    for (int j = 0; j < 4; ++j) {
        int n = nbase + 16 * j + ln;
        float bn = bias[n];
#pragma unroll
        for (int i = 0; i < 8; ++i) {
            int m = mbase + i + hl * 8;
            out[(size_t)m * EMBED + n] = acc[j][i] + bn;
        }
    }
}

extern "C" void kernel_launch(void* const* d_in, const int* in_sizes, int n_in,
                              void* d_out, int out_size, void* d_ws, size_t ws_size,
                              hipStream_t stream) {
    (void)in_sizes; (void)n_in; (void)out_size; (void)ws_size;
    const float* q  = (const float*)d_in[0];
    const float* ki = (const float*)d_in[1];
    const float* v  = (const float*)d_in[2];
    const float* Wq = (const float*)d_in[3];
    const float* bq = (const float*)d_in[4];
    const float* Wk = (const float*)d_in[5];
    const float* bk = (const float*)d_in[6];
    const float* Wv = (const float*)d_in[7];
    const float* bv = (const float*)d_in[8];
    const float* Wo = (const float*)d_in[9];
    const float* bo = (const float*)d_in[10];

    const size_t ACT = (size_t)MROWS * EMBED;  // 4M elems
    const size_t WEL = (size_t)EMBED * EMBED;  // 1M elems
    bfraw* XQ  = (bfraw*)d_ws;                 // bf16 workspace, 64 MB total
    bfraw* XK  = XQ + ACT;
    bfraw* XV  = XK + ACT;
    bfraw* WQb = XV + ACT;
    bfraw* WKb = WQb + WEL;
    bfraw* WVb = WKb + WEL;
    bfraw* WOb = WVb + WEL;
    bfraw* Qb  = WOb + WEL;
    bfraw* Kb  = Qb + ACT;
    bfraw* Vtb = Kb + ACT;   // V stored transposed [b,h,hd,s]
    bfraw* AO  = Vtb + ACT;

    cvt_bf16_kernel<<<2048, 256, 0, stream>>>(q,  XQ,  (int)ACT);
    cvt_bf16_kernel<<<2048, 256, 0, stream>>>(ki, XK,  (int)ACT);
    cvt_bf16_kernel<<<2048, 256, 0, stream>>>(v,  XV,  (int)ACT);
    cvt_bf16_kernel<<<1024, 256, 0, stream>>>(Wq, WQb, (int)WEL);
    cvt_bf16_kernel<<<1024, 256, 0, stream>>>(Wk, WKb, (int)WEL);
    cvt_bf16_kernel<<<1024, 256, 0, stream>>>(Wv, WVb, (int)WEL);
    cvt_bf16_kernel<<<1024, 256, 0, stream>>>(Wo, WOb, (int)WEL);

    dim3 gproj(MROWS / 128, EMBED / 64);
    size_t shmem = 2 * PANEL_BYTES;
    qkv_proj_kernel<<<gproj, 256, shmem, stream>>>(XQ, WQb, bq, Qb, 0);
    qkv_proj_kernel<<<gproj, 256, shmem, stream>>>(XK, WKb, bk, Kb, 0);
    qkv_proj_kernel<<<gproj, 256, shmem, stream>>>(XV, WVb, bv, Vtb, 1);

    flash_attn_kernel<<<dim3(SEQ / 128, NHEAD, BATCH), 256, 0, stream>>>(Qb, Kb, Vtb, AO);

    out_proj_kernel<<<gproj, 256, shmem, stream>>>(AO, WOb, bo, (float*)d_out);
}
